// MVGFRNN_44968307589654
// MI455X (gfx1250) — compile-verified, matrix operand order checked by reading, not run on hardware
//
#include <hip/hip_runtime.h>
#include <math.h>

#define B_ 8
#define V_ 3
#define N_ 400
#define S_ 1200
#define F_ 16
#define H_ 32
#define K_ 10
#define STEPS_ 12
#define L_ (V_*S_)            // 3600 rows of x
#define NEG_ (-9.0e15f)
#define ALPHA_ 0.8f
#define QROWS_ 48             // padded query rows per batch (3 views * 16)
#define QV_ 11                // valid rows per view: K labels + 1 unlabeled

typedef _Float16 h16 __attribute__((ext_vector_type(16)));
typedef float    f8  __attribute__((ext_vector_type(8)));
typedef unsigned int u32x4 __attribute__((ext_vector_type(4)));
typedef int          i32x4 __attribute__((ext_vector_type(4)));
typedef int          i32x8 __attribute__((ext_vector_type(8)));

union H16 { h16 v; _Float16 e[16]; };
union F8  { f8  v; float    e[8];  };

// element j of a lane's A/B fragment maps to K-index (16-bit 16x32 WMMA layout)
__device__ __forceinline__ int kmap(int j, int hi) {
  return (j < 8 ? j : j + 8) + (hi ? 8 : 0);
}
__device__ __forceinline__ f8 wmma_f16(h16 a, h16 b, f8 c) {
  return __builtin_amdgcn_wmma_f32_16x16x32_f16(false, a, false, b, (short)0, c, false, false);
}
__device__ __forceinline__ float sigmoidf_(float x){ return 1.f/(1.f+__expf(-x)); }

// ---------------------------------------------------------------- K1: Wh = feat @ W_gat (per b,v)
__global__ void wh_kernel(const float* __restrict__ feat, const float* __restrict__ Wg,
                          float* __restrict__ Wh) {
  int wave = (blockIdx.x*blockDim.x + threadIdx.x) >> 5;        // 1800 waves
  int sb = wave % (S_/16); int v = (wave/(S_/16)) % V_; int b = wave/((S_/16)*V_);
  int l = threadIdx.x & 31, r = l & 15, hi = l >> 4;
  int s = sb*16 + r;
  H16 a;
  #pragma unroll
  for (int j=0;j<16;++j){ int kk=kmap(j,hi);
    a.e[j] = (kk<F_) ? (_Float16)feat[((size_t)b*S_+s)*F_+kk] : (_Float16)0.f; }
  #pragma unroll
  for (int n=0;n<2;++n){
    H16 bm; int h = n*16 + r;
    #pragma unroll
    for (int j=0;j<16;++j){ int kk=kmap(j,hi);
      bm.e[j] = (kk<F_) ? (_Float16)Wg[((size_t)v*F_+kk)*H_+h] : (_Float16)0.f; }
    F8 c = {};
    c.v = wmma_f16(a.v, bm.v, c.v);
    #pragma unroll
    for (int rp=0;rp<8;++rp){
      int row = sb*16 + rp + hi*8;
      Wh[(((size_t)b*V_+v)*S_ + row)*H_ + h] = c.e[rp];
    }
  }
}

// ---------------------------------------------------------------- K1b: e1/e2 row dots (lane = h)
__global__ void e_kernel(const float* __restrict__ Wh, const float* __restrict__ ag,
                         float* __restrict__ e1, float* __restrict__ e2) {
  int wave = (blockIdx.x*blockDim.x + threadIdx.x) >> 5;        // 28800 rows
  int l = threadIdx.x & 31;
  int v = (wave / S_) % V_;
  float w  = Wh[(size_t)wave*H_ + l];
  float p1 = w * ag[v*2*H_ + l];
  float p2 = w * ag[v*2*H_ + H_ + l];
  #pragma unroll
  for (int m=16;m>=1;m>>=1){ p1 += __shfl_xor(p1,m,32); p2 += __shfl_xor(p2,m,32); }
  if (l==0){ e1[wave]=p1; e2[wave]=p2; }
}

// ---------------------------------------------------------------- K2: GAT softmax stats (online)
__global__ void gat_stats_kernel(const float* __restrict__ adj, const float* __restrict__ e1,
                                 const float* __restrict__ e2,
                                 float* __restrict__ mrow, float* __restrict__ lrow) {
  int wave = (blockIdx.x*blockDim.x + threadIdx.x) >> 5;        // 28800 rows
  int l = threadIdx.x & 31;
  int bv = wave / S_;
  const float* arow = adj + (size_t)wave * S_;
  const float* e2p  = e2  + (size_t)bv * S_;
  float e1r = e1[wave];
  float m = -INFINITY, ls = 0.f;
  for (int t=l; t<S_; t+=32){
    float aij = arow[t];
    float xv  = e1r + e2p[t];
    float val = (aij>0.f) ? (xv>0.f ? xv : 0.2f*xv) : NEG_;
    if (val > m){ ls = ls*__expf(m-val) + 1.f; m = val; }
    else        { ls += __expf(val-m); }
  }
  #pragma unroll
  for (int msk=16;msk>=1;msk>>=1){
    float m2=__shfl_xor(m,msk,32), l2=__shfl_xor(ls,msk,32);
    float M = fmaxf(m,m2);
    ls = ls*__expf(m-M) + l2*__expf(m2-M);
    m = M;
  }
  if (l==0){ mrow[wave]=m; lrow[wave]=ls; }
}

// ---------------------------------------------------------------- K3: hid = elu(att @ Wh) via WMMA
// Blocks of 4 waves share one (b,v); Wh chunks are staged to LDS by the
// Tensor Data Mover (double-buffered, TENSORcnt-paced), cutting L2 traffic 4x.
__global__ void gat_agg_kernel(const float* __restrict__ adj, const float* __restrict__ e1,
                               const float* __restrict__ e2, const float* __restrict__ mrow,
                               const float* __restrict__ lrow, const float* __restrict__ Wh,
                               float* __restrict__ xout) {
  __shared__ float whbuf[2][32][H_];                            // 8 KB TDM staging
  int bv = blockIdx.y;                                          // 0..23 = b*V + v
  int v  = bv % V_; int b = bv / V_;
  int wid = threadIdx.x >> 5;
  int sb = blockIdx.x*4 + wid; if (sb > S_/16 - 1) sb = S_/16 - 1;   // clamp phantom tile
  int l = threadIdx.x & 31, r = l & 15, hi = l >> 4;
  int s = sb*16 + r;
  int rowi = bv*S_ + s;
  float e1r = e1[rowi], mr = mrow[rowi], invl = 1.f/lrow[rowi];
  const float* arow = adj + (size_t)rowi * S_;
  const float* e2p  = e2  + (size_t)bv * S_;
  const float* WhP  = Wh  + (size_t)bv * S_ * H_;
  bool issuer = (wid == 0);

  const int NCH = (S_ + 31) / 32;                               // 38 chunks

  // Issue one TDM load: 32x32 f32 tile of Wh[b,v] rows [t0, t0+32) -> whbuf[buf]
  auto tdm_issue = [&](int chunk){
    int t0 = chunk * 32;
    unsigned long long ga = (unsigned long long)(const void*)(WhP + (size_t)t0*H_);
    unsigned int lds_off  = (unsigned int)(size_t)&whbuf[chunk & 1][0][0];
    int td1 = S_ - t0;                                          // remaining rows (OOB reads -> 0)
    u32x4 g0;
    g0[0] = 1u;                                                 // count=1, user descriptor
    g0[1] = lds_off;                                            // lds_addr (bytes)
    g0[2] = (unsigned int)ga;                                   // global_addr[31:0]
    g0[3] = ((unsigned int)(ga >> 32) & 0x01FFFFFFu) | (2u << 30); // addr[56:32] | type=2
    i32x8 g1;
    g1[0] = (2 << 16);                                          // data_size = 4 bytes
    g1[1] = (H_ << 16);                                         // tensor_dim0 = 32 (lo16)
    g1[2] = ((td1 & 0xFFFF) << 16);                             // tensor_dim1 lo16
    g1[3] = ((td1 >> 16) & 0xFFFF) | (32 << 16);                // tensor_dim1 hi | tile_dim0=32
    g1[4] = 32;                                                 // tile_dim1=32, tile_dim2=0
    g1[5] = H_;                                                 // tensor_dim0_stride = 32 elems
    g1[6] = 0;
    g1[7] = 0;
    i32x4 z4 = {0,0,0,0};                                       // groups 2/3 unused (2-D tensor)
    i32x8 z8 = {0,0,0,0,0,0,0,0};                               // extra group (6-arg toolchain)
    __builtin_amdgcn_tensor_load_to_lds(g0, g1, z4, z4, z8, 0);
  };

  if (issuer) tdm_issue(0);
  F8 c0 = {}, c1 = {};
  for (int ch=0; ch<NCH; ++ch){
    if (issuer){
      if (ch+1 < NCH){ tdm_issue(ch+1); __builtin_amdgcn_s_wait_tensorcnt(1); }
      else           {                  __builtin_amdgcn_s_wait_tensorcnt(0); }
    }
    __syncthreads();                                            // whbuf[ch&1] ready
    int t0 = ch*32;
    if (t0+32 < S_) __builtin_prefetch(arow + t0 + 32, 0, 0);
    H16 a;
    #pragma unroll
    for (int j=0;j<16;++j){
      int t = t0 + kmap(j,hi);
      float p = 0.f;
      if (t < S_){
        float aij = arow[t];
        float xv  = e1r + e2p[t];
        float val = (aij>0.f) ? (xv>0.f ? xv : 0.2f*xv) : NEG_;
        p = __expf(val - mr) * invl;
      }
      a.e[j] = (_Float16)p;
    }
    #pragma unroll
    for (int n=0;n<2;++n){
      H16 bm; int h = n*16 + r;
      #pragma unroll
      for (int j=0;j<16;++j)
        bm.e[j] = (_Float16)whbuf[ch & 1][kmap(j,hi)][h];       // LDS, TDM zero-fills tail
      if (n==0) c0.v = wmma_f16(a.v, bm.v, c0.v);
      else      c1.v = wmma_f16(a.v, bm.v, c1.v);
    }
    __syncthreads();                                            // done consuming before reuse
  }
  #pragma unroll
  for (int n=0;n<2;++n){
    F8& c = n ? c1 : c0; int h = n*16 + r;
    #pragma unroll
    for (int rp=0;rp<8;++rp){
      int row = sb*16 + rp + hi*8;
      float val = c.e[rp];
      val = val>0.f ? val : expm1f(val);                        // elu
      xout[(((size_t)b)*L_ + v*S_ + row)*H_ + h] = val;
    }
  }
}

// ---------------------------------------------------------------- K4: Q/K/V projection (6 WMMAs/tile)
__global__ void qkv_kernel(const float* __restrict__ x, const float* __restrict__ Wq,
                           const float* __restrict__ bq,
                           float* __restrict__ qm, float* __restrict__ km, float* __restrict__ vm) {
  int wave = (blockIdx.x*blockDim.x + threadIdx.x) >> 5;        // 1800 tiles over B*L
  int l = threadIdx.x & 31, r = l & 15, hi = l >> 4;
  int g0 = wave*16;
  H16 a;
  #pragma unroll
  for (int j=0;j<16;++j) a.e[j] = (_Float16)x[((size_t)(g0 + r))*H_ + kmap(j,hi)];
  #pragma unroll
  for (int n=0;n<6;++n){
    H16 bm; int g = n*16 + r;                                   // output col 0..95
    #pragma unroll
    for (int j=0;j<16;++j) bm.e[j] = (_Float16)Wq[(size_t)kmap(j,hi)*96 + g];
    F8 c = {};
    c.v = wmma_f16(a.v, bm.v, c.v);
    float bias = bq[g];
    float* dst = (g < 32) ? qm : (g < 64 ? km : vm);
    int col = g & 31;
    #pragma unroll
    for (int rp=0;rp<8;++rp){
      int row = g0 + rp + hi*8;
      dst[(size_t)row*H_ + col] = c.e[rp] + bias;
    }
  }
}

// ---------------------------------------------------------------- K5: SA softmax stats for 33 q-rows/b
__global__ void sa_stats_kernel(const float* __restrict__ qm, const float* __restrict__ km,
                                float* __restrict__ am, float* __restrict__ al) {
  int wave = (blockIdx.x*blockDim.x + threadIdx.x) >> 5;        // B*48 = 384 waves
  int l = threadIdx.x & 31;
  int b = wave / QROWS_; int p = wave % QROWS_;
  int v = p >> 4; int i = p & 15; if (i > 10) i = 10;           // pad rows dup row 10
  int lr = v*S_ + 2*N_ + i;
  const float* q = qm + ((size_t)b*L_ + lr)*H_;
  float qr[32];
  #pragma unroll
  for (int j=0;j<32;++j) qr[j] = q[j];
  const float scale = 0.17677669529663687f;                     // 1/sqrt(H)
  float m = -INFINITY, ls = 0.f;
  for (int key=l; key<L_; key+=32){
    const float* kp = km + ((size_t)b*L_ + key)*H_;
    float s = 0.f;
    #pragma unroll
    for (int j=0;j<32;++j) s += qr[j]*kp[j];
    s *= scale;
    if (s > m){ ls = ls*__expf(m-s) + 1.f; m = s; } else ls += __expf(s-m);
  }
  #pragma unroll
  for (int msk=16;msk>=1;msk>>=1){
    float m2=__shfl_xor(m,msk,32), l2=__shfl_xor(ls,msk,32);
    float M = fmaxf(m,m2);
    ls = ls*__expf(m-M) + l2*__expf(m2-M);
    m = M;
  }
  if (l==0){ am[wave]=m; al[wave]=ls; }
}

// ---------------------------------------------------------------- K6: SA + o/fuse/gate, all WMMA
__global__ void sa_apply_kernel(const float* __restrict__ qm, const float* __restrict__ km,
                                const float* __restrict__ vm, const float* __restrict__ x,
                                const float* __restrict__ am, const float* __restrict__ al,
                                const float* __restrict__ oW, const float* __restrict__ ob,
                                const float* __restrict__ fW, const float* __restrict__ fb,
                                float* __restrict__ gh) {
  __shared__ _Float16 lds[4][16][32];                           // per-wave C->A transpose staging
  int wid = threadIdx.x >> 5;
  int wave = blockIdx.x*4 + wid;                                // 24 tiles (8 b * 3 views)
  int b = wave / 3; int tile = wave % 3; int p0 = tile*16;
  int l = threadIdx.x & 31, r = l & 15, hi = l >> 4;
  auto lrow_of = [](int p){ int i = p & 15; if (i > 10) i = 10; int v = p >> 4; return v*S_ + 2*N_ + i; };

  int lra = lrow_of(p0 + r);
  H16 aq;
  #pragma unroll
  for (int j=0;j<16;++j) aq.e[j] = (_Float16)qm[((size_t)b*L_ + lra)*H_ + kmap(j,hi)];
  float m8[8], il8[8];
  #pragma unroll
  for (int rp=0;rp<8;++rp){
    int pp = p0 + rp + hi*8;
    m8[rp] = am[b*QROWS_ + pp];
    il8[rp] = 1.f/al[b*QROWS_ + pp];
  }
  const float scale = 0.17677669529663687f;
  F8 o0 = {}, o1 = {};
  for (int k0=0; k0<L_; k0+=32){
    F8 sc[2];
    #pragma unroll
    for (int n=0;n<2;++n){
      H16 bk; int key = k0 + n*16 + r;
      #pragma unroll
      for (int j=0;j<16;++j)
        bk.e[j] = (key < L_) ? (_Float16)km[((size_t)b*L_ + key)*H_ + kmap(j,hi)] : (_Float16)0.f;
      F8 z = {};
      sc[n].v = wmma_f16(aq.v, bk.v, z.v);
    }
    // P = exp(S*scale - m)/l  -> LDS (C layout -> A layout transpose)
    #pragma unroll
    for (int n=0;n<2;++n){
      int key = k0 + n*16 + r;
      #pragma unroll
      for (int rp=0;rp<8;++rp){
        float pv = 0.f;
        if (key < L_) pv = __expf(sc[n].e[rp]*scale - m8[rp]) * il8[rp];
        lds[wid][rp + hi*8][n*16 + r] = (_Float16)pv;
      }
    }
    __builtin_amdgcn_wave_barrier();
    asm volatile("s_wait_dscnt 0" ::: "memory");
    H16 ap;
    #pragma unroll
    for (int j=0;j<16;++j) ap.e[j] = lds[wid][r][kmap(j,hi)];
    #pragma unroll
    for (int n=0;n<2;++n){
      H16 bv; int h = n*16 + r;
      #pragma unroll
      for (int j=0;j<16;++j){
        int key = k0 + kmap(j,hi);
        bv.e[j] = (key < L_) ? (_Float16)vm[((size_t)b*L_ + key)*H_ + h] : (_Float16)0.f;
      }
      if (n==0) o0.v = wmma_f16(ap.v, bv.v, o0.v);
      else      o1.v = wmma_f16(ap.v, bv.v, o1.v);
    }
    __builtin_amdgcn_wave_barrier();
  }
  // sa -> A layout -> project with o_W
  #pragma unroll
  for (int n=0;n<2;++n){
    F8& c = n ? o1 : o0;
    #pragma unroll
    for (int rp=0;rp<8;++rp) lds[wid][rp + hi*8][n*16 + r] = (_Float16)c.e[rp];
  }
  __builtin_amdgcn_wave_barrier();
  asm volatile("s_wait_dscnt 0" ::: "memory");
  H16 as;
  #pragma unroll
  for (int j=0;j<16;++j) as.e[j] = lds[wid][r][kmap(j,hi)];
  F8 so[2];
  #pragma unroll
  for (int n=0;n<2;++n){
    H16 bo; int h = n*16 + r;
    #pragma unroll
    for (int j=0;j<16;++j) bo.e[j] = (_Float16)oW[(size_t)kmap(j,hi)*H_ + h];
    F8 z = {};
    so[n].v = wmma_f16(as.v, bo.v, z.v);
  }
  __builtin_amdgcn_wave_barrier();
  // hfuse = 0.8*(sa@oW + ob) + 0.2*x ; keep in regs + stage to LDS for fuse matmul
  float hf[2][8];
  #pragma unroll
  for (int n=0;n<2;++n){
    int h = n*16 + r; float bias = ob[h];
    #pragma unroll
    for (int rp=0;rp<8;++rp){
      int pp = p0 + rp + hi*8; int lr2 = lrow_of(pp);
      float xv = x[((size_t)b*L_ + lr2)*H_ + h];
      float v  = ALPHA_*(so[n].e[rp] + bias) + (1.f-ALPHA_)*xv;
      hf[n][rp] = v;
      lds[wid][rp + hi*8][n*16 + r] = (_Float16)v;
    }
  }
  __builtin_amdgcn_wave_barrier();
  asm volatile("s_wait_dscnt 0" ::: "memory");
  H16 ah;
  #pragma unroll
  for (int j=0;j<16;++j) ah.e[j] = lds[wid][r][kmap(j,hi)];
  #pragma unroll
  for (int n=0;n<2;++n){
    H16 bf; int h = n*16 + r;
    #pragma unroll
    for (int j=0;j<16;++j) bf.e[j] = (_Float16)fW[(size_t)kmap(j,hi)*H_ + h];
    F8 z = {};
    F8 g; g.v = wmma_f16(ah.v, bf.v, z.v);
    float bias = fb[h];
    #pragma unroll
    for (int rp=0;rp<8;++rp){
      float gate = sigmoidf_(g.e[rp] + bias);
      int pp = p0 + rp + hi*8;
      gh[((size_t)b*QROWS_ + pp)*H_ + h] = gate * hf[n][rp];
    }
  }
}

// ---------------------------------------------------------------- K7: fusion + label attention + GRU
__global__ void head_kernel(const float* __restrict__ gh, const float* __restrict__ dis,
                            const float* __restrict__ a1W, const float* __restrict__ a1b,
                            const float* __restrict__ a2W, const float* __restrict__ a2b,
                            const float* __restrict__ Wih, const float* __restrict__ Whh,
                            const float* __restrict__ bih, const float* __restrict__ bhh,
                            const float* __restrict__ oW, const float* __restrict__ ob2,
                            float* __restrict__ out) {
  int b = blockIdx.x;
  int l = threadIdx.x;                                          // 32 threads, lane = h
  __shared__ float fl[QV_][H_];
  for (int i=0;i<QV_;++i){
    float f = 0.f;
    #pragma unroll
    for (int v=0; v<V_; ++v) f += gh[((size_t)b*QROWS_ + v*16 + i)*H_ + l];
    fl[i][l] = f;
  }
  __syncthreads();
  float s10[K_];
  for (int i=0;i<K_;++i){
    float acc = a1b[l];
    for (int j=0;j<H_;++j){
      acc += fl[i][j]     * a1W[j*H_ + l];
      acc += fl[QV_-1][j] * a1W[(H_+j)*H_ + l];
    }
    acc = fmaxf(acc, 0.f);
    float sv = acc * a2W[l];
    #pragma unroll
    for (int m=16;m>=1;m>>=1) sv += __shfl_xor(sv,m,32);
    s10[i] = (sv + a2b[0]) * dis[b*K_ + i];
  }
  float mx = -INFINITY;
  for (int i=0;i<K_;++i) mx = fmaxf(mx, s10[i]);
  float den = 0.f;
  for (int i=0;i<K_;++i){ s10[i] = __expf(s10[i]-mx); den += s10[i]; }
  float ctx = 0.f;
  for (int i=0;i<K_;++i) ctx += (s10[i]/den) * fl[i][l];
  float xt = ctx, hst = fl[QV_-1][l];
  for (int t=0;t<STEPS_;++t){
    float ir=bih[l], iz=bih[H_+l], in=bih[2*H_+l];
    float hr=bhh[l], hz=bhh[H_+l], hn=bhh[2*H_+l];
    for (int j=0;j<H_;++j){
      float xj = __shfl(xt, j, 32), hj = __shfl(hst, j, 32);
      ir += xj*Wih[j*3*H_ + l];       iz += xj*Wih[j*3*H_ + H_ + l];   in += xj*Wih[j*3*H_ + 2*H_ + l];
      hr += hj*Whh[j*3*H_ + l];       hz += hj*Whh[j*3*H_ + H_ + l];   hn += hj*Whh[j*3*H_ + 2*H_ + l];
    }
    float rg = sigmoidf_(ir+hr);
    float zg = sigmoidf_(iz+hz);
    float ng = tanhf(in + rg*hn);
    float h2 = (1.f-zg)*ng + zg*hst;
    float o = h2 * oW[l];
    #pragma unroll
    for (int m=16;m>=1;m>>=1) o += __shfl_xor(o,m,32);
    if (l==0) out[b*STEPS_ + t] = o + ob2[0];
    xt = h2; hst = h2;
  }
}

extern "C" void kernel_launch(void* const* d_in, const int* in_sizes, int n_in,
                              void* d_out, int out_size, void* d_ws, size_t ws_size,
                              hipStream_t stream) {
  const float* feat = (const float*)d_in[0];
  const float* adj  = (const float*)d_in[1];
  const float* dis  = (const float*)d_in[2];
  const float* Wg   = (const float*)d_in[3];
  const float* ag   = (const float*)d_in[4];
  const float* qkvW = (const float*)d_in[5];
  const float* qkvb = (const float*)d_in[6];
  const float* oW   = (const float*)d_in[7];
  const float* ob   = (const float*)d_in[8];
  const float* fW   = (const float*)d_in[9];
  const float* fb   = (const float*)d_in[10];
  const float* a1W  = (const float*)d_in[11];
  const float* a1b  = (const float*)d_in[12];
  const float* a2W  = (const float*)d_in[13];
  const float* a2b  = (const float*)d_in[14];
  const float* Wih  = (const float*)d_in[15];
  const float* Whh  = (const float*)d_in[16];
  const float* bih  = (const float*)d_in[17];
  const float* bhh  = (const float*)d_in[18];
  const float* outW = (const float*)d_in[19];
  const float* outb = (const float*)d_in[20];

  float* ws = (float*)d_ws;
  size_t o = 0;
  float* Wh = ws + o; o += (size_t)B_*V_*S_*H_;   // 921600
  float* e1 = ws + o; o += (size_t)B_*V_*S_;      // 28800
  float* e2 = ws + o; o += (size_t)B_*V_*S_;
  float* mr = ws + o; o += (size_t)B_*V_*S_;
  float* lr = ws + o; o += (size_t)B_*V_*S_;
  float* x  = ws + o; o += (size_t)B_*L_*H_;      // 921600 each
  float* qm = ws + o; o += (size_t)B_*L_*H_;
  float* km = ws + o; o += (size_t)B_*L_*H_;
  float* vm = ws + o; o += (size_t)B_*L_*H_;
  float* am = ws + o; o += (size_t)B_*QROWS_;
  float* al = ws + o; o += (size_t)B_*QROWS_;
  float* gh = ws + o; o += (size_t)B_*QROWS_*H_;

  wh_kernel       <<<450, 128, 0, stream>>>(feat, Wg, Wh);
  e_kernel        <<<7200,128, 0, stream>>>(Wh, ag, e1, e2);
  gat_stats_kernel<<<7200,128, 0, stream>>>(adj, e1, e2, mr, lr);
  gat_agg_kernel  <<<dim3(19,24), 128, 0, stream>>>(adj, e1, e2, mr, lr, Wh, x);
  qkv_kernel      <<<450, 128, 0, stream>>>(x, qkvW, qkvb, qm, km, vm);
  sa_stats_kernel <<<96,  128, 0, stream>>>(qm, km, am, al);
  sa_apply_kernel <<<6,   128, 0, stream>>>(qm, km, vm, x, am, al, oW, ob, fW, fb, gh);
  head_kernel     <<<B_,  32,  0, stream>>>(gh, dis, a1W, a1b, a2W, a2b, Wih, Whh, bih, bhh, outW, outb, (float*)d_out);
}
